// Attention_15616501088603
// MI455X (gfx1250) — matrix operand order repaired and checked by reading, not compiled, not run on hardware
//
#include <hip/hip_runtime.h>
#include <hip/hip_bf16.h>
#include <stdint.h>

// ---------------------------------------------------------------------------
// CDNA5 (gfx1250) attention pipeline: bf16 WMMA + TDM/async LDS staging.
// ---------------------------------------------------------------------------

typedef __attribute__((ext_vector_type(16))) __bf16 v16bf;
typedef __attribute__((ext_vector_type(8)))  float  v8f;

union FragU {
    v16bf v;
    uint4 q[2];
};

#define LDS_AS __attribute__((address_space(3)))
#define GLB_AS __attribute__((address_space(1)))

#if defined(__gfx1250__) && __has_builtin(__builtin_amdgcn_tensor_load_to_lds)
#define HAVE_TDM 1
#else
#define HAVE_TDM 0
#endif

#if defined(__gfx1250__) && __has_builtin(__builtin_amdgcn_global_load_async_to_lds_b128)
#define HAVE_ASYNC 1
#else
#define HAVE_ASYNC 0
#endif

__device__ __forceinline__ uint16_t f2bf(float f) {
    uint32_t u = __float_as_uint(f);
    uint32_t r = (u + 0x7FFFu + ((u >> 16) & 1u)) >> 16;
    return (uint16_t)r;
}

#if HAVE_TDM
typedef unsigned int u32x4 __attribute__((ext_vector_type(4)));
typedef int          i32x4 __attribute__((ext_vector_type(4)));
typedef int          i32x8 __attribute__((ext_vector_type(8)));

// Issue one TDM load of a 128-row x 32-bf16 tile (row stride K elements) into
// LDS at byte offset lds_off, padding each 64B of data with 16B so LDS rows
// land on an 80B stride (matches the ds_load_b128 fragment addressing).
__device__ __forceinline__ void tdm_load_tile(const uint16_t* gtile,
                                              uint32_t lds_off,
                                              int K, int rows_total)
{
    unsigned long long ga = (unsigned long long)(uintptr_t)gtile;
    u32x4 g0;
    g0[0] = 1u;                                   // count = 1 valid descriptor
    g0[1] = lds_off;                              // lds_addr (bytes)
    g0[2] = (unsigned)(ga & 0xFFFFFFFFu);         // global_addr[31:0]
    g0[3] = (unsigned)((ga >> 32) & 0x1FFFFFFu)   // global_addr[56:32]
          | (2u << 30);                           // type = 2 ("image")
    i32x8 g1;
    g1[0] = (1 << 16)                             // data_size = 2 bytes
          | (1 << 20)                             // pad_enable
          | (3 << 22)                             // pad_interval: 16 DWORDs (64B)
          | (3 << 25);                            // pad_amount:   4 DWORDs (16B)
    g1[1] = (int)((K & 0xFFFF) << 16);            // tensor_dim0[15:0]
    g1[2] = (int)(((rows_total & 0xFFFF) << 16)   // tensor_dim1[15:0]
          | ((K >> 16) & 0xFFFF));                // tensor_dim0[31:16]
    g1[3] = (int)(32 << 16);                      // tile_dim0 = 32 elements
    g1[4] = 128;                                  // tile_dim1 = 128 rows
    g1[5] = K;                                    // tensor_dim0_stride[31:0]
    g1[6] = 0;
    g1[7] = 0;
    i32x4 z4 = {0, 0, 0, 0};
#if __clang_major__ >= 23
    i32x8 z8 = {0, 0, 0, 0, 0, 0, 0, 0};
    __builtin_amdgcn_tensor_load_to_lds(g0, g1, z4, z4, z8, 0);
#else
    __builtin_amdgcn_tensor_load_to_lds(g0, g1, z4, z4, 0);
#endif
}
#endif  // HAVE_TDM

__device__ __forceinline__ uint32_t lds_byte_offset(const void* p) {
#if defined(__gfx1250__)
    return (uint32_t)(uintptr_t)(LDS_AS const void*)p;
#else
    (void)p;
    return 0;
#endif
}

// Stage one 128x32 bf16 tile of A and of B into LDS (80B padded rows).
__device__ __forceinline__ void stage_tiles(
    const uint16_t* __restrict__ A, const uint16_t* __restrict__ B,
    uint16_t* As, uint16_t* Bs,
    long blockM, long blockN, int K, int M, int N, int k0, int t)
{
#if HAVE_TDM
    (void)M; (void)N;
    if (t < 32) {  // wave 0 drives the Tensor Data Mover
        tdm_load_tile(A + blockM * (long)K + k0, lds_byte_offset(As), K, M);
        tdm_load_tile(B + blockN * (long)K + k0, lds_byte_offset(Bs), K, N);
    }
#elif HAVE_ASYNC
    (void)M; (void)N;
    const int lrow = t >> 1, lhalf = t & 1;
    const uint16_t* gA = A + (blockM + lrow) * (long)K + k0 + lhalf * 16;
    const uint16_t* gB = B + (blockN + lrow) * (long)K + k0 + lhalf * 16;
    uint16_t* lA = As + lrow * 40 + lhalf * 16;
    uint16_t* lB = Bs + lrow * 40 + lhalf * 16;
    __builtin_amdgcn_global_load_async_to_lds_b128((GLB_AS void*)(void*)gA,
                                                   (LDS_AS void*)(void*)lA, 0, 0);
    __builtin_amdgcn_global_load_async_to_lds_b128((GLB_AS void*)(void*)gA,
                                                   (LDS_AS void*)(void*)lA, 16, 0);
    __builtin_amdgcn_global_load_async_to_lds_b128((GLB_AS void*)(void*)gB,
                                                   (LDS_AS void*)(void*)lB, 0, 0);
    __builtin_amdgcn_global_load_async_to_lds_b128((GLB_AS void*)(void*)gB,
                                                   (LDS_AS void*)(void*)lB, 16, 0);
#else
    (void)M; (void)N;
    const int lrow = t >> 1, lhalf = t & 1;
    const uint4* srcA = (const uint4*)(A + (blockM + lrow) * (long)K + k0 + lhalf * 16);
    const uint4* srcB = (const uint4*)(B + (blockN + lrow) * (long)K + k0 + lhalf * 16);
    uint4 a0 = srcA[0], a1 = srcA[1];
    uint4 b0 = srcB[0], b1 = srcB[1];
    uint4* dstA = (uint4*)(As + lrow * 40 + lhalf * 16);
    uint4* dstB = (uint4*)(Bs + lrow * 40 + lhalf * 16);
    dstA[0] = a0; dstA[1] = a1;
    dstB[0] = b0; dstB[1] = b1;
#endif
}

// Wait for staged tile to land, then workgroup barrier.
__device__ __forceinline__ void stage_wait(int t)
{
#if HAVE_TDM
    if (t < 32) {
#if __has_builtin(__builtin_amdgcn_s_wait_tensorcnt)
        __builtin_amdgcn_s_wait_tensorcnt(0);
#else
        asm volatile("s_wait_tensorcnt 0x0" ::: "memory");
#endif
    }
#elif HAVE_ASYNC
#if __has_builtin(__builtin_amdgcn_s_wait_asynccnt)
    __builtin_amdgcn_s_wait_asynccnt(0);
#else
    asm volatile("s_wait_asynccnt 0x0" ::: "memory");
#endif
#else
    (void)t;
#endif
    __syncthreads();
}

// ---------------------------------------------------------------------------
// Tiled NT GEMM: C[M,N] = A[M,K] * B[N,K]^T, A/B bf16 (as uint16), C fp32.
// Block tile 128x128, 256 threads = 8 waves (4 M x 2 N), wave tile 32x64.
// Double-buffered LDS; staging via TDM / async-to-LDS / sync fallback.
// M,N multiples of 128; K multiple of 32 (true for all calls here).
// ---------------------------------------------------------------------------
__global__ __launch_bounds__(256) void gemm_bf16_nt(
    const uint16_t* __restrict__ A,
    const uint16_t* __restrict__ B,
    float* __restrict__ C,
    int M, int N, int K)
{
    // 128 rows x 32 bf16, padded to 40 bf16 (80 B) per row; two buffers.
    __shared__ uint16_t As[2][128 * 40];
    __shared__ uint16_t Bs[2][128 * 40];

    const int t    = threadIdx.x;
    const int wave = t >> 5;
    const int lane = t & 31;
    const int waveM = wave & 3;   // 0..3  (rows of 32)
    const int waveN = wave >> 2;  // 0..1  (cols of 64)
    const int lr15  = lane & 15;
    const int hi    = lane >> 4;  // 0: K 0-7,16-23; 1: K 8-15,24-31

    const long blockM = (long)blockIdx.y * 128;
    const long blockN = (long)blockIdx.x * 128;

    const v8f vzero = {0.f, 0.f, 0.f, 0.f, 0.f, 0.f, 0.f, 0.f};
    v8f acc[2][4];
#pragma unroll
    for (int mi = 0; mi < 2; ++mi)
#pragma unroll
        for (int ni = 0; ni < 4; ++ni)
            acc[mi][ni] = vzero;

    // prologue: stage K-step 0 into buffer 0
    stage_tiles(A, B, &As[0][0], &Bs[0][0], blockM, blockN, K, M, N, 0, t);

    int buf = 0;
    for (int k0 = 0; k0 < K; k0 += 32) {
        stage_wait(t);  // current buffer resident; previous compute drained

        if (k0 + 32 < K)
            stage_tiles(A, B, &As[buf ^ 1][0], &Bs[buf ^ 1][0],
                        blockM, blockN, K, M, N, k0 + 32, t);

        const uint4* As4 = (const uint4*)&As[buf][0];
        const uint4* Bs4 = (const uint4*)&Bs[buf][0];

        FragU a[2], b[4];
#pragma unroll
        for (int mi = 0; mi < 2; ++mi) {
            const int r = waveM * 32 + mi * 16 + lr15;
            a[mi].q[0] = As4[r * 5 + hi];       // K pairs {kb..kb+7}, kb = 8*hi
            a[mi].q[1] = As4[r * 5 + hi + 2];   // K pairs {kb+16..kb+23}
        }
#pragma unroll
        for (int ni = 0; ni < 4; ++ni) {
            const int c = waveN * 64 + ni * 16 + lr15;
            b[ni].q[0] = Bs4[c * 5 + hi];       // K {kb..kb+7}, kb = 8*hi
            b[ni].q[1] = Bs4[c * 5 + hi + 2];   // K {kb+16..kb+23}
        }

#pragma unroll
        for (int mi = 0; mi < 2; ++mi)
#pragma unroll
            for (int ni = 0; ni < 4; ++ni)
                acc[mi][ni] = __builtin_amdgcn_wmma_f32_16x16x32_bf16(
                    false, a[mi].v, false, b[ni].v,
                    (short)0, acc[mi][ni], false, false);

        buf ^= 1;
    }

    // epilogue: C/D layout (VGPR r: M = r + 8*hi, N = lane&15)
#pragma unroll
    for (int mi = 0; mi < 2; ++mi)
#pragma unroll
        for (int ni = 0; ni < 4; ++ni) {
            const long row0 = blockM + waveM * 32 + mi * 16 + hi * 8;
            const long col  = blockN + waveN * 64 + ni * 16 + lr15;
#pragma unroll
            for (int r = 0; r < 8; ++r)
                C[(row0 + r) * (long)N + col] = acc[mi][ni][r];
        }
}

// ---------------------------------------------------------------------------
// fp32 -> bf16 elementwise convert (grid-stride)
// ---------------------------------------------------------------------------
__global__ void cvt_f32_bf16(const float* __restrict__ in,
                             uint16_t* __restrict__ out, long n)
{
    long i = (long)blockIdx.x * blockDim.x + threadIdx.x;
    const long stride = (long)gridDim.x * blockDim.x;
    for (; i < n; i += stride) out[i] = f2bf(in[i]);
}

// ---------------------------------------------------------------------------
// Per-batch transpose + convert: out[z][c][r] = bf16(in[z][r][c])
// block (32,8), grid (cols/32, rows/32, batches)
// ---------------------------------------------------------------------------
__global__ __launch_bounds__(256) void transpose_cvt(
    const float* __restrict__ in, uint16_t* __restrict__ out,
    int rows, int cols)
{
    __shared__ float tile[32][33];
    const long base = (long)blockIdx.z * rows * cols;
    const int tx = threadIdx.x;
    const int ty = threadIdx.y;
    const int r0 = blockIdx.y * 32;
    const int c0 = blockIdx.x * 32;
#pragma unroll
    for (int i = 0; i < 32; i += 8)
        tile[ty + i][tx] = in[base + (long)(r0 + ty + i) * cols + c0 + tx];
    __syncthreads();
#pragma unroll
    for (int i = 0; i < 32; i += 8)
        out[base + (long)(c0 + ty + i) * rows + r0 + tx] = f2bf(tile[tx][ty + i]);
}

// ---------------------------------------------------------------------------
// Row softmax with scale folded in; fp32 in, bf16 out. One block per row.
// ---------------------------------------------------------------------------
__global__ __launch_bounds__(256) void softmax_scale_bf16(
    const float* __restrict__ S, uint16_t* __restrict__ P,
    int ncols, float scale)
{
    const float* row = S + (long)blockIdx.x * ncols;
    uint16_t* prow   = P + (long)blockIdx.x * ncols;
    __shared__ float red[256];
    const int t = threadIdx.x;

    float vals[16];
    float vmax = -INFINITY;
#pragma unroll
    for (int i = 0; i < 16; ++i) {
        float v = row[t + i * 256] * scale;
        vals[i] = v;
        vmax = fmaxf(vmax, v);
    }
    red[t] = vmax;
    __syncthreads();
    for (int s = 128; s > 0; s >>= 1) {
        if (t < s) red[t] = fmaxf(red[t], red[t + s]);
        __syncthreads();
    }
    const float m = red[0];
    __syncthreads();

    float sum = 0.f;
#pragma unroll
    for (int i = 0; i < 16; ++i) {
        float e = __expf(vals[i] - m);
        vals[i] = e;
        sum += e;
    }
    red[t] = sum;
    __syncthreads();
    for (int s = 128; s > 0; s >>= 1) {
        if (t < s) red[t] += red[t + s];
        __syncthreads();
    }
    const float inv = 1.0f / red[0];
#pragma unroll
    for (int i = 0; i < 16; ++i)
        prow[t + i * 256] = f2bf(vals[i] * inv);
    (void)ncols;
}

// ---------------------------------------------------------------------------
// Orchestration
// ---------------------------------------------------------------------------
extern "C" void kernel_launch(void* const* d_in, const int* in_sizes, int n_in,
                              void* d_out, int out_size, void* d_ws, size_t ws_size,
                              hipStream_t stream)
{
    (void)in_sizes; (void)n_in; (void)out_size; (void)ws_size;

    const long DIM  = 2048;
    const long SEQ  = 4096;
    const long BAT  = 2;
    const long MTOT = BAT * SEQ;          // 8192
    const float scale = 0.022097086912079608f;  // 1/sqrt(2048)

    const float* x  = (const float*)d_in[0];
    const float* wq = (const float*)d_in[1];
    const float* wk = (const float*)d_in[2];
    const float* wv = (const float*)d_in[3];
    const float* wo = (const float*)d_in[4];
    float* out = (float*)d_out;

    char* ws = (char*)d_ws;
    const size_t MB = 1ull << 20;
    uint16_t* xb    = (uint16_t*)(ws + 0 * MB);     // 32 MB  bf16 x
    uint16_t* wqb   = (uint16_t*)(ws + 32 * MB);    //  8 MB
    uint16_t* wkb   = (uint16_t*)(ws + 40 * MB);    //  8 MB
    uint16_t* wvb   = (uint16_t*)(ws + 48 * MB);    //  8 MB
    uint16_t* wob   = (uint16_t*)(ws + 56 * MB);    //  8 MB
    uint16_t* Qb    = (uint16_t*)(ws + 64 * MB);    // 32 MB
    uint16_t* Kb    = (uint16_t*)(ws + 96 * MB);    // 32 MB
    uint16_t* Vt    = (uint16_t*)(ws + 128 * MB);   // 32 MB  [b][d][k]
    float*    tmpf  = (float*)   (ws + 160 * MB);   // 64 MB  Qf/Kf/Vf then Ctxf
    float*    Sf    = (float*)   (ws + 224 * MB);   // 64 MB  per-batch scores
    uint16_t* attnb = (uint16_t*)(ws + 288 * MB);   // 32 MB  per-batch probs
    uint16_t* ctxb  = (uint16_t*)(ws + 224 * MB);   // 32 MB  (reuses Sf region)

    // 1) downconvert inputs to bf16
    cvt_f32_bf16<<<4096, 256, 0, stream>>>(x,  xb,  MTOT * DIM);
    cvt_f32_bf16<<<1024, 256, 0, stream>>>(wq, wqb, DIM * DIM);
    cvt_f32_bf16<<<1024, 256, 0, stream>>>(wk, wkb, DIM * DIM);
    cvt_f32_bf16<<<1024, 256, 0, stream>>>(wv, wvb, DIM * DIM);
    cvt_f32_bf16<<<1024, 256, 0, stream>>>(wo, wob, DIM * DIM);

    const dim3 gProj((unsigned)(DIM / 128), (unsigned)(MTOT / 128));  // 16 x 64

    // 2) Q = x Wq^T ; K = x Wk^T (kept as bf16 for QK^T)
    gemm_bf16_nt<<<gProj, 256, 0, stream>>>(xb, wqb, tmpf, (int)MTOT, (int)DIM, (int)DIM);
    cvt_f32_bf16<<<4096, 256, 0, stream>>>(tmpf, Qb, MTOT * DIM);
    gemm_bf16_nt<<<gProj, 256, 0, stream>>>(xb, wkb, tmpf, (int)MTOT, (int)DIM, (int)DIM);
    cvt_f32_bf16<<<4096, 256, 0, stream>>>(tmpf, Kb, MTOT * DIM);

    // 3) V = x Wv^T, then per-batch transpose to [d][k] bf16 for NT PV GEMM
    gemm_bf16_nt<<<gProj, 256, 0, stream>>>(xb, wvb, tmpf, (int)MTOT, (int)DIM, (int)DIM);
    {
        dim3 tg((unsigned)(DIM / 32), (unsigned)(SEQ / 32), (unsigned)BAT);
        dim3 tb(32, 8);
        transpose_cvt<<<tg, tb, 0, stream>>>(tmpf, Vt, (int)SEQ, (int)DIM);
    }

    // 4) per batch: S = Q K^T, P = softmax(S/sqrt(d)), Ctx = P V
    for (long b = 0; b < BAT; ++b) {
        const uint16_t* Qbb = Qb + b * SEQ * DIM;
        const uint16_t* Kbb = Kb + b * SEQ * DIM;
        const uint16_t* Vtb = Vt + b * SEQ * DIM;
        float* Ctxb_f = tmpf + b * SEQ * DIM;

        dim3 gS((unsigned)(SEQ / 128), (unsigned)(SEQ / 128));  // 32 x 32
        gemm_bf16_nt<<<gS, 256, 0, stream>>>(Qbb, Kbb, Sf, (int)SEQ, (int)SEQ, (int)DIM);

        softmax_scale_bf16<<<(unsigned)SEQ, 256, 0, stream>>>(Sf, attnb, (int)SEQ, scale);

        dim3 gC((unsigned)(DIM / 128), (unsigned)(SEQ / 128));  // 16 x 32
        gemm_bf16_nt<<<gC, 256, 0, stream>>>(attnb, Vtb, Ctxb_f, (int)SEQ, (int)DIM, (int)SEQ);
    }

    // 5) out = Ctx Wo^T
    cvt_f32_bf16<<<4096, 256, 0, stream>>>(tmpf, ctxb, MTOT * DIM);
    gemm_bf16_nt<<<gProj, 256, 0, stream>>>(ctxb, wob, out, (int)MTOT, (int)DIM, (int)DIM);
}
